// DynSMHALayer_69544110457492
// MI455X (gfx1250) — matrix operand order repaired and checked by reading, not compiled, not run on hardware
//
#include <hip/hip_runtime.h>
#include <cstdint>
#include <cstddef>

#define NTOK 8192
#define CDIM 2048
#define HDIM 128
#define NEXP 16
#define BDIM 4
#define TDIM 2048

typedef __attribute__((ext_vector_type(16))) __bf16 v16bf;
typedef __attribute__((ext_vector_type(8)))  float  v8f;
typedef __attribute__((ext_vector_type(4)))  unsigned int vu4;
typedef __attribute__((ext_vector_type(4)))  int v4i;

#ifndef __has_builtin
#define __has_builtin(x) 0
#endif

#define HAVE_ASYNC_LDS __has_builtin(__builtin_amdgcn_global_load_async_to_lds_b128)

union Frag {
    v16bf v;
    unsigned short u[16];
    vu4 q[2];
};

__device__ __forceinline__ unsigned short f2bf(float f) {
    union { float f; unsigned u; } x; x.f = f;
    unsigned r = x.u + 0x7FFFu + ((x.u >> 16) & 1u);
    return (unsigned short)(r >> 16);
}

__device__ __forceinline__ v8f wmma_bf16(v16bf a, v16bf b, v8f c) {
    return __builtin_amdgcn_wmma_f32_16x16x32_bf16(
        false, a, false, b, (short)0, c, false, false);
}

// 16-byte global -> LDS copy; async (no VGPR staging) when the toolchain has it.
__device__ __forceinline__ void copy16(const unsigned short* g, unsigned short* l) {
#if HAVE_ASYNC_LDS
    __builtin_amdgcn_global_load_async_to_lds_b128((v4i*)g, (v4i*)l, 0, 0);
#else
    *(vu4*)l = *(const vu4*)g;
#endif
}

// Wait for this wave's outstanding async LDS copies (no-op on fallback path).
__device__ __forceinline__ void async_wait() {
#if HAVE_ASYNC_LDS
#if __has_builtin(__builtin_amdgcn_s_wait_asynccnt)
    __builtin_amdgcn_s_wait_asynccnt(0);
#else
    asm volatile("s_wait_asynccnt 0x0" ::: "memory");
#endif
#endif
}

// A-matrix fragment (16x32 bf16, M x K), source row-major [M][K].
// lane<16: M=lane, K = k0 + {0..7, 16..23}; lane>=16: K = k0 + {8..15, 24..31}.
__device__ __forceinline__ v16bf load_a(const unsigned short* base, int ld, int k0) {
    int lane = threadIdx.x & 31;
    int m = lane & 15, hi = lane >> 4;
    const unsigned short* p = base + (size_t)m * ld + k0 + 8 * hi;
    Frag f;
    f.q[0] = *(const vu4*)(p);
    f.q[1] = *(const vu4*)(p + 16);
    return f.v;
}

// B-matrix fragment (32x16 bf16, K x N), source N-major rows, K contiguous:
// value(K,n) = base[n*ld + k0 + K].  lane<16: K = k0 + {0..7, 16..23}; lane>=16: K = k0 + {8..15, 24..31}.
__device__ __forceinline__ v16bf load_b_kmajor(const unsigned short* base, int ld, int k0) {
    int lane = threadIdx.x & 31;
    int n = lane & 15, hi = lane >> 4;
    const unsigned short* p = base + (size_t)n * ld + k0 + 8 * hi;
    Frag f;
    f.q[0] = *(const vu4*)(p);
    f.q[1] = *(const vu4*)(p + 16);
    return f.v;
}

// ---------------- kernel 1: column norms of sim_matrix ----------------
__global__ void simnorm_kernel(const float* __restrict__ sim, float* __restrict__ sn) {
    int e = threadIdx.x;
    if (e < NEXP) {
        float s = 0.f;
        for (int c = 0; c < CDIM; ++c) {
            float v = sim[(size_t)c * NEXP + e];
            s += v * v;
        }
        sn[e] = fmaxf(sqrtf(s), 1e-12f);
    }
}

// ---------------- kernel 2: gating -> w[N,E] ----------------
__global__ void gating_kernel(const float* __restrict__ x, const float* __restrict__ sim,
                              const float* __restrict__ gates, const float* __restrict__ sn,
                              float* __restrict__ w) {
    int wave = threadIdx.x >> 5, lane = threadIdx.x & 31;
    int n = blockIdx.x * 8 + wave;
    float dot[NEXP];
#pragma unroll
    for (int e = 0; e < NEXP; ++e) dot[e] = 0.f;
    float xn = 0.f;
    for (int c = lane; c < CDIM; c += 32) {
        float xv = x[(size_t)n * CDIM + c];
        xn += xv * xv;
        const float* sr = sim + (size_t)c * NEXP;
#pragma unroll
        for (int e = 0; e < NEXP; ++e) dot[e] += xv * sr[e];
    }
#pragma unroll
    for (int s = 1; s < 32; s <<= 1) {
        xn += __shfl_xor(xn, s, 32);
#pragma unroll
        for (int e = 0; e < NEXP; ++e) dot[e] += __shfl_xor(dot[e], s, 32);
    }
    float xnrm = fmaxf(sqrtf(xn), 1e-12f);
    int e = lane & 15;
    float d = 0.f;
#pragma unroll
    for (int k = 0; k < NEXP; ++k)
        if (e == k) d = dot[k];
    float sg = 1.f / (1.f + __expf(-gates[e]));
    float logit = d / (xnrm * sn[e]) - sg;
    float gated = fmaxf(logit, 0.f);
    int active = logit > 0.f;
    unsigned long long bal = __ballot(active && (lane < 16));
    bool inactive = ((unsigned)bal & 0xFFFFu) == 0u;
    float l1 = logit; int i1 = e;
#pragma unroll
    for (int s = 1; s < 16; s <<= 1) {
        float ol = __shfl_xor(l1, s, 32);
        int oi = __shfl_xor(i1, s, 32);
        if (ol > l1 || (ol == l1 && oi < i1)) { l1 = ol; i1 = oi; }
    }
    float l2 = (e == i1) ? -3.0e38f : logit; int i2 = e;
#pragma unroll
    for (int s = 1; s < 16; s <<= 1) {
        float ol = __shfl_xor(l2, s, 32);
        int oi = __shfl_xor(i2, s, 32);
        if (ol > l2 || (ol == l2 && oi < i2)) { l2 = ol; i2 = oi; }
    }
    float mask = active ? 1.f : 0.f;
    if (inactive && (e == i1 || e == i2)) mask = 1.f;
    float ml = (mask > 0.f) ? gated : -1e9f;
    float mx = ml;
#pragma unroll
    for (int s = 1; s < 16; s <<= 1) mx = fmaxf(mx, __shfl_xor(mx, s, 32));
    float p = __expf(ml - mx);
    float sum = p;
#pragma unroll
    for (int s = 1; s < 16; s <<= 1) sum += __shfl_xor(sum, s, 32);
    float wt = (mask > 0.f) ? p / sum : 0.f;
    if (lane < 16) w[(size_t)n * NEXP + lane] = wt;
}

// ---------------- kernel 3a: f32 -> bf16 (straight) ----------------
__global__ void cvt_bf16_kernel(const float* __restrict__ s, unsigned short* __restrict__ d, int n) {
    for (int i = blockIdx.x * blockDim.x + threadIdx.x; i < n; i += gridDim.x * blockDim.x)
        d[i] = f2bf(s[i]);
}

// ---------------- kernel 3b: f32 [E][R][S] -> bf16 transposed [E][S][R] ----------------
__global__ void cvt_tr_bf16_kernel(const float* __restrict__ s, unsigned short* __restrict__ d,
                                   int R, int S, int total) {
    for (int i = blockIdx.x * blockDim.x + threadIdx.x; i < total; i += gridDim.x * blockDim.x) {
        int e = i / (R * S);
        int rem = i - e * (R * S);
        int r = rem / S;
        int c = rem - r * S;
        d[((size_t)e * S + c) * R + r] = f2bf(s[i]);
    }
}

// ---------------- kernel 4: mixed-expert QKV projection ----------------
// out[n,h] = sum_e w[n,e] * (x[n,:] @ proj[e,:,h]); proj supplied transposed [E][H][C].
// V result is written transposed [H][NTOK] for the attention P@V B-operand.
__global__ void __launch_bounds__(256)
qkv_kernel(const unsigned short* __restrict__ xb,
           const unsigned short* __restrict__ qbT,
           const unsigned short* __restrict__ kbT,
           const unsigned short* __restrict__ vbT,
           const float* __restrict__ wbuf,
           unsigned short* __restrict__ qt,
           unsigned short* __restrict__ kt,
           unsigned short* __restrict__ vtT) {
    __shared__ unsigned short xs[128 * 32];   // A tile: [row][K]
    __shared__ unsigned short ps[128 * 32];   // B tile (K-major): [h][K]
    const unsigned short* pb = (blockIdx.y == 0) ? qbT : ((blockIdx.y == 1) ? kbT : vbT);
    int wave = threadIdx.x >> 5, lane = threadIdx.x & 31;
    int m = lane & 15, hi = lane >> 4;
    int rowbase = blockIdx.x * 128;

    v8f acc[8] = {};
    for (int e = 0; e < NEXP; ++e) {
        v8f pacc[8] = {};
        for (int kc = 0; kc < CDIM / 32; ++kc) {
            int k0 = kc * 32;
            __syncthreads();
            {   // x tile: 128 rows x 32 K
                int i = threadIdx.x * 16;
                int r = i >> 5, cc = i & 31;
                const unsigned short* g = xb + (size_t)(rowbase + r) * CDIM + k0 + cc;
                copy16(g, &xs[r * 32 + cc]);
                copy16(g + 8, &xs[r * 32 + cc + 8]);
            }
            {   // proj tile K-major: 128 h x 32 K, from projT[e][h][k0+k]
                int i = threadIdx.x * 16;
                int r = i >> 5, cc = i & 31;
                const unsigned short* g = pb + ((size_t)e * HDIM + r) * CDIM + k0 + cc;
                copy16(g, &ps[r * 32 + cc]);
                copy16(g + 8, &ps[r * 32 + cc + 8]);
            }
            async_wait();
            __syncthreads();
            v16bf a = load_a(xs + wave * 16 * 32, 32, 0);
#pragma unroll
            for (int ct = 0; ct < 8; ++ct) {
                v16bf b = load_b_kmajor(ps + ct * 16 * 32, 32, 0);
                pacc[ct] = wmma_bf16(a, b, pacc[ct]);
            }
        }
#pragma unroll
        for (int j = 0; j < 8; ++j) {
            int row = rowbase + wave * 16 + j + 8 * hi;
            float wj = wbuf[(size_t)row * NEXP + e];
#pragma unroll
            for (int ct = 0; ct < 8; ++ct) acc[ct][j] += wj * pacc[ct][j];
        }
    }
    if (blockIdx.y == 2) {
#pragma unroll
        for (int ct = 0; ct < 8; ++ct)
#pragma unroll
            for (int j = 0; j < 8; ++j) {
                int row = rowbase + wave * 16 + j + 8 * hi;
                vtT[(size_t)(ct * 16 + m) * NTOK + row] = f2bf(acc[ct][j]);
            }
    } else {
        unsigned short* op = (blockIdx.y == 0) ? qt : kt;
#pragma unroll
        for (int ct = 0; ct < 8; ++ct)
#pragma unroll
            for (int j = 0; j < 8; ++j) {
                int row = rowbase + wave * 16 + j + 8 * hi;
                op[(size_t)row * HDIM + ct * 16 + m] = f2bf(acc[ct][j]);
            }
    }
}

// ---------------- kernel 5: causal flash attention ----------------
__global__ void __launch_bounds__(256)
attn_kernel(const unsigned short* __restrict__ qt,
            const unsigned short* __restrict__ kt,
            const unsigned short* __restrict__ vtT,
            unsigned short* __restrict__ aob) {
    __shared__ unsigned short ks[128 * 128];  // K tile [key][h]; reused for P staging
    __shared__ unsigned short vs[128 * 128];  // V tile transposed [h][key]
    int wave = threadIdx.x >> 5, lane = threadIdx.x & 31;
    int m = lane & 15, hi = lane >> 4;
    int qb0 = blockIdx.x * 128;
    size_t nbase = (size_t)blockIdx.y * TDIM;
    const float scale = 0.08838834764831845f;  // 1/sqrt(128)

    v16bf aq[4];
#pragma unroll
    for (int s4 = 0; s4 < 4; ++s4)
        aq[s4] = load_a(qt + (nbase + qb0 + wave * 16) * HDIM, HDIM, s4 * 32);

    v8f oacc[8] = {};
    float mrow[8], lrow[8];
#pragma unroll
    for (int j = 0; j < 8; ++j) { mrow[j] = -1e30f; lrow[j] = 0.f; }

    for (int jb = 0; jb <= (int)blockIdx.x; ++jb) {
        __syncthreads();
        {   // K tile [key][h] from kt; V tile [h][key] from vtT
            int i = threadIdx.x * 64;
            int r = i >> 7, c0 = i & 127;
            const unsigned short* gk = kt + (nbase + jb * 128 + r) * HDIM + c0;
            const unsigned short* gv = vtT + (size_t)r * NTOK + nbase + jb * 128 + c0;
#pragma unroll
            for (int q = 0; q < 8; ++q) {
                copy16(gk + q * 8, &ks[r * 128 + c0 + q * 8]);
                copy16(gv + q * 8, &vs[r * 128 + c0 + q * 8]);
            }
        }
        async_wait();
        __syncthreads();
        // S = Q @ K^T (wave: 16 q-rows x 128 keys)
        v8f st[8] = {};
#pragma unroll
        for (int ct = 0; ct < 8; ++ct)
#pragma unroll
            for (int s4 = 0; s4 < 4; ++s4) {
                v16bf bk = load_b_kmajor(ks + ct * 16 * 128, 128, s4 * 32);
                st[ct] = wmma_bf16(aq[s4], bk, st[ct]);
            }
        // scale + causal mask
#pragma unroll
        for (int ct = 0; ct < 8; ++ct)
#pragma unroll
            for (int j = 0; j < 8; ++j) {
                int qrow = qb0 + wave * 16 + j + 8 * hi;
                int key = jb * 128 + ct * 16 + m;
                float v = st[ct][j] * scale;
                if (key > qrow) v = -1e9f;
                st[ct][j] = v;
            }
        // online softmax per row j
#pragma unroll
        for (int j = 0; j < 8; ++j) {
            float rm = st[0][j];
#pragma unroll
            for (int ct = 1; ct < 8; ++ct) rm = fmaxf(rm, st[ct][j]);
#pragma unroll
            for (int s = 1; s < 16; s <<= 1) rm = fmaxf(rm, __shfl_xor(rm, s, 32));
            float mn = fmaxf(mrow[j], rm);
            float corr = __expf(mrow[j] - mn);
            mrow[j] = mn;
            lrow[j] *= corr;
#pragma unroll
            for (int ht = 0; ht < 8; ++ht) oacc[ht][j] *= corr;
            float rs = 0.f;
#pragma unroll
            for (int ct = 0; ct < 8; ++ct) {
                float p = __expf(st[ct][j] - mn);
                st[ct][j] = p;
                rs += p;
            }
#pragma unroll
            for (int s = 1; s < 16; s <<= 1) rs += __shfl_xor(rs, s, 32);
            lrow[j] += rs;
        }
        // stage P (bf16) into ks area (K tile no longer needed)
        __syncthreads();
#pragma unroll
        for (int ct = 0; ct < 8; ++ct)
#pragma unroll
            for (int j = 0; j < 8; ++j)
                ks[(wave * 16 + j + 8 * hi) * 128 + ct * 16 + m] = f2bf(st[ct][j]);
        __syncthreads();
        // O += P @ V  (V B-operand K-major from transposed tile)
#pragma unroll
        for (int s4 = 0; s4 < 4; ++s4) {
            v16bf ap = load_a(ks + wave * 16 * 128, 128, s4 * 32);
#pragma unroll
            for (int ht = 0; ht < 8; ++ht) {
                v16bf bv = load_b_kmajor(vs + ht * 16 * 128, 128, s4 * 32);
                oacc[ht] = wmma_bf16(ap, bv, oacc[ht]);
            }
        }
    }
#pragma unroll
    for (int j = 0; j < 8; ++j) {
        float inv = 1.f / lrow[j];
        size_t row = nbase + qb0 + wave * 16 + j + 8 * hi;
#pragma unroll
        for (int ht = 0; ht < 8; ++ht)
            aob[row * HDIM + ht * 16 + m] = f2bf(oacc[ht][j] * inv);
    }
}

// ---------------- kernel 6: mixed-expert output projection ----------------
// out[n,c] = sum_e w[n,e] * (ao[n,:] @ o_proj[e,:,c]); o_proj supplied transposed [E][C][H].
__global__ void __launch_bounds__(256)
outproj_kernel(const unsigned short* __restrict__ aob,
               const unsigned short* __restrict__ obT,
               const float* __restrict__ wbuf,
               float* __restrict__ out) {
    __shared__ unsigned short as2[128 * 128];  // ao tile [row][h]
    __shared__ unsigned short bs[128 * 128];   // o_proj tile K-major [c][h]
    int wave = threadIdx.x >> 5, lane = threadIdx.x & 31;
    int m = lane & 15, hi = lane >> 4;
    int colbase = blockIdx.x * 128;
    int rowbase = blockIdx.y * 128;

    {   // ao tile once: 128 rows x H=128
        int i = threadIdx.x * 64;
        int r = i >> 7, c0 = i & 127;
        const unsigned short* g = aob + (size_t)(rowbase + r) * HDIM + c0;
#pragma unroll
        for (int q = 0; q < 8; ++q)
            copy16(g + q * 8, &as2[r * 128 + c0 + q * 8]);
    }
    async_wait();
    __syncthreads();
    v16bf afr[4];
#pragma unroll
    for (int s4 = 0; s4 < 4; ++s4)
        afr[s4] = load_a(as2 + wave * 16 * 128, 128, s4 * 32);

    v8f acc[8] = {};
    for (int e = 0; e < NEXP; ++e) {
        __syncthreads();
        {   // obT tile: [c][h], 128 x 128, rows contiguous in h
            int i = threadIdx.x * 64;
            int r = i >> 7, c0 = i & 127;
            const unsigned short* g = obT + ((size_t)e * CDIM + colbase + r) * HDIM + c0;
#pragma unroll
            for (int q = 0; q < 8; ++q)
                copy16(g + q * 8, &bs[r * 128 + c0 + q * 8]);
        }
        async_wait();
        __syncthreads();
        v8f pacc[8] = {};
#pragma unroll
        for (int s4 = 0; s4 < 4; ++s4)
#pragma unroll
            for (int ct = 0; ct < 8; ++ct) {
                v16bf b = load_b_kmajor(bs + ct * 16 * 128, 128, s4 * 32);
                pacc[ct] = wmma_bf16(afr[s4], b, pacc[ct]);
            }
#pragma unroll
        for (int j = 0; j < 8; ++j) {
            int row = rowbase + wave * 16 + j + 8 * hi;
            float wj = wbuf[(size_t)row * NEXP + e];
#pragma unroll
            for (int ct = 0; ct < 8; ++ct) acc[ct][j] += wj * pacc[ct][j];
        }
    }
#pragma unroll
    for (int ct = 0; ct < 8; ++ct)
#pragma unroll
        for (int j = 0; j < 8; ++j) {
            size_t row = rowbase + wave * 16 + j + 8 * hi;
            out[row * CDIM + colbase + ct * 16 + m] = acc[ct][j];
        }
}

extern "C" void kernel_launch(void* const* d_in, const int* in_sizes, int n_in,
                              void* d_out, int out_size, void* d_ws, size_t ws_size,
                              hipStream_t stream) {
    const float* x     = (const float*)d_in[0];
    const float* sim   = (const float*)d_in[1];
    const float* gates = (const float*)d_in[2];
    const float* qp    = (const float*)d_in[3];
    const float* kp    = (const float*)d_in[4];
    const float* vp    = (const float*)d_in[5];
    const float* opj   = (const float*)d_in[6];

    char* ws = (char*)d_ws;
    size_t off = 0;
    auto walloc = [&](size_t bytes) -> void* {
        void* p = ws + off;
        off += (bytes + 255) & ~(size_t)255;
        return p;
    };
    float* simnorm = (float*)walloc(NEXP * 4);
    float* wbuf    = (float*)walloc((size_t)NTOK * NEXP * 4);
    unsigned short* xb  = (unsigned short*)walloc((size_t)NTOK * CDIM * 2);
    unsigned short* qbT = (unsigned short*)walloc((size_t)NEXP * CDIM * HDIM * 2);
    unsigned short* kbT = (unsigned short*)walloc((size_t)NEXP * CDIM * HDIM * 2);
    unsigned short* vbT = (unsigned short*)walloc((size_t)NEXP * CDIM * HDIM * 2);
    unsigned short* obT = (unsigned short*)walloc((size_t)NEXP * HDIM * CDIM * 2);
    unsigned short* qt  = (unsigned short*)walloc((size_t)NTOK * HDIM * 2);
    unsigned short* kt  = (unsigned short*)walloc((size_t)NTOK * HDIM * 2);
    unsigned short* vtT = (unsigned short*)walloc((size_t)NTOK * HDIM * 2);
    unsigned short* aob = (unsigned short*)walloc((size_t)NTOK * HDIM * 2);

    simnorm_kernel<<<1, 64, 0, stream>>>(sim, simnorm);
    gating_kernel<<<NTOK / 8, 256, 0, stream>>>(x, sim, gates, simnorm, wbuf);

    cvt_bf16_kernel<<<2048, 256, 0, stream>>>(x, xb, NTOK * CDIM);
    // q/k/v: [E][C][H] -> [E][H][C]
    cvt_tr_bf16_kernel<<<2048, 256, 0, stream>>>(qp, qbT, CDIM, HDIM, NEXP * CDIM * HDIM);
    cvt_tr_bf16_kernel<<<2048, 256, 0, stream>>>(kp, kbT, CDIM, HDIM, NEXP * CDIM * HDIM);
    cvt_tr_bf16_kernel<<<2048, 256, 0, stream>>>(vp, vbT, CDIM, HDIM, NEXP * CDIM * HDIM);
    // o: [E][H][C] -> [E][C][H]
    cvt_tr_bf16_kernel<<<2048, 256, 0, stream>>>(opj, obT, HDIM, CDIM, NEXP * HDIM * CDIM);

    qkv_kernel<<<dim3(NTOK / 128, 3), 256, 0, stream>>>(xb, qbT, kbT, vbT, wbuf, qt, kt, vtT);
    attn_kernel<<<dim3(TDIM / 128, BDIM), 256, 0, stream>>>(qt, kt, vtT, aob);
    outproj_kernel<<<dim3(CDIM / 128, NTOK / 128), 256, 0, stream>>>(aob, obT, wbuf, (float*)d_out);
}
